// BiMambaBlock_11433202942758
// MI455X (gfx1250) — matrix operand order repaired and checked by reading, not compiled, not run on hardware
//
#include <hip/hip_runtime.h>
#include <hip/hip_bf16.h>
#include <stdint.h>

// ---------------------------------------------------------------------------
// BiMamba block for MI455X (gfx1250, wave32, WMMA).
// GEMMs: bf16 x bf16 -> f32 accumulate via v_wmma_f32_16x16x32_bf16,
//        register-blocked 64x32 per wave (8 WMMA per K-step, 12 b128 loads).
// Scan:  fp32 VALU, state in registers, one wave per (batch, head, dir).
// ---------------------------------------------------------------------------

#define D_MODEL   768
#define D_STATE   32
#define D_CONV    4
#define D_INNER   1536
#define NHEADS    24
#define HEADDIM   64
#define CONV_DIM  1600            // D_INNER + 2*D_STATE
#define D_IN_PROJ 3160            // 2*D_INNER + 2*D_STATE + NHEADS
#define D_IN_PAD  3168            // padded to multiple of 32 for WMMA blocks
#define B_SZ      2
#define SEQ       512
#define NTOK      (B_SZ*SEQ)      // 1024
#define EPS       1e-5f

typedef __bf16 bf16_t;
typedef __attribute__((ext_vector_type(16))) __bf16 v16bf;
typedef __attribute__((ext_vector_type(8)))  __bf16 v8bf;
typedef __attribute__((ext_vector_type(8)))  float  v8f;

// ---------------- helpers ----------------

__device__ inline unsigned short f2bf(float f) {
  unsigned int u = __float_as_uint(f);
  u += 0x7fffu + ((u >> 16) & 1u);           // round-to-nearest-even
  return (unsigned short)(u >> 16);
}

__device__ inline float sigmoidf_(float v) { return 1.f / (1.f + expf(-v)); }

__device__ inline float block_reduce_sum(float v, float* red) {
  #pragma unroll
  for (int o = 16; o > 0; o >>= 1) v += __shfl_xor(v, o, 32);
  int wid = threadIdx.x >> 5, lid = threadIdx.x & 31;
  if (lid == 0) red[wid] = v;
  __syncthreads();
  int nw = blockDim.x >> 5;
  if (wid == 0) {
    float w = (lid < nw) ? red[lid] : 0.f;
    #pragma unroll
    for (int o = 16; o > 0; o >>= 1) w += __shfl_xor(w, o, 32);
    if (lid == 0) red[0] = w;
  }
  __syncthreads();
  return red[0];
}

// ---------------- kernel 1: RMSNorm(x) -> bf16 h ----------------

__global__ __launch_bounds__(256) void rmsnorm_in_kernel(
    const float* __restrict__ x, const float* __restrict__ w,
    unsigned short* __restrict__ out) {
  __shared__ float red[32];
  int t = blockIdx.x;
  const float* xr = x + (size_t)t * D_MODEL;
  float ss = 0.f;
  for (int i = threadIdx.x; i < D_MODEL; i += 256) { float v = xr[i]; ss += v * v; }
  ss = block_reduce_sum(ss, red);
  float r = rsqrtf(ss / (float)D_MODEL + EPS);
  for (int i = threadIdx.x; i < D_MODEL; i += 256)
    out[(size_t)t * D_MODEL + i] = f2bf(xr[i] * r * w[i]);
}

// ---------------- weight conversion ----------------

__global__ __launch_bounds__(256) void cvt_bf16_kernel(
    const float* __restrict__ src, unsigned short* __restrict__ dst, int n) {
  int i = blockIdx.x * 256 + threadIdx.x;
  if (i < n) dst[i] = f2bf(src[i]);
}

// pad rows [rows_src, rows_dst) with zeros
__global__ __launch_bounds__(256) void cvt_bf16_pad_kernel(
    const float* __restrict__ src, unsigned short* __restrict__ dst,
    int rows_src, int rows_dst, int cols) {
  int i = blockIdx.x * 256 + threadIdx.x;
  int n = rows_dst * cols;
  if (i >= n) return;
  int r = i / cols, c = i % cols;
  dst[i] = (r < rows_src) ? f2bf(src[(size_t)r * cols + c]) : (unsigned short)0;
}

// ---------------- WMMA GEMM: C[M,N] = A[M,K](bf16) * W[N,K](bf16)^T (+addend)
// Register-blocked: each wave computes a 64x32 output block
// (MT=4 x NT=2 tiles of 16x16), K stepped by 32.
// Per K-step: 4 A-fragments + 2 B-fragments (12 b128 loads) -> 8 WMMAs.

#define GEMM_MT 4
#define GEMM_NT 2

__global__ __launch_bounds__(128) void gemm_bf16_wmma_kernel(
    const bf16_t* __restrict__ A, int lda,
    const bf16_t* __restrict__ W, int ldw,
    float* __restrict__ C, int ldc,
    const float* __restrict__ addend,
    int M, int N, int K) {
  int lane = threadIdx.x & 31;
  int wave = threadIdx.x >> 5;
  int blocksN = N / (16 * GEMM_NT);
  int blocks  = (M / (16 * GEMM_MT)) * blocksN;
  int blk = blockIdx.x * 4 + wave;            // wave-uniform
  if (blk >= blocks) return;
  int tmBase = (blk / blocksN) * (16 * GEMM_MT);
  int tnBase = (blk % blocksN) * (16 * GEMM_NT);

  // A fragment (16x32 bf16): lanes 0-15 row=lane,    K 0..15;
  //                          lanes 16-31 row=lane-16, K 16..31.
  const bf16_t* pa[GEMM_MT];
  #pragma unroll
  for (int mi = 0; mi < GEMM_MT; ++mi)
    pa[mi] = A + (size_t)(tmBase + mi * 16 + (lane & 15)) * lda + ((lane < 16) ? 0 : 16);
  // B fragment (32x16 bf16): lanes 0-15 col=lane,    K 0..15;
  //                          lanes 16-31 col=lane-16, K 16..31.
  const bf16_t* pw[GEMM_NT];
  #pragma unroll
  for (int ni = 0; ni < GEMM_NT; ++ni)
    pw[ni] = W + (size_t)(tnBase + ni * 16 + (lane & 15)) * ldw + ((lane < 16) ? 0 : 16);

  v8f acc[GEMM_MT][GEMM_NT];
  #pragma unroll
  for (int mi = 0; mi < GEMM_MT; ++mi)
    #pragma unroll
    for (int ni = 0; ni < GEMM_NT; ++ni)
      acc[mi][ni] = v8f{};

  for (int k = 0; k < K; k += 32) {
    __builtin_prefetch((const void*)(pa[0] + k + 128), 0, 3);
    __builtin_prefetch((const void*)(pw[0] + k + 128), 0, 3);
    v16bf af[GEMM_MT], bf[GEMM_NT];
    #pragma unroll
    for (int mi = 0; mi < GEMM_MT; ++mi) {
      v8bf a0 = *(const v8bf*)(pa[mi] + k);
      v8bf a1 = *(const v8bf*)(pa[mi] + k + 8);
      #pragma unroll
      for (int i = 0; i < 8; ++i) { af[mi][i] = a0[i]; af[mi][8 + i] = a1[i]; }
    }
    #pragma unroll
    for (int ni = 0; ni < GEMM_NT; ++ni)
      bf[ni] = *(const v16bf*)(pw[ni] + k);
    #pragma unroll
    for (int mi = 0; mi < GEMM_MT; ++mi)
      #pragma unroll
      for (int ni = 0; ni < GEMM_NT; ++ni)
        acc[mi][ni] = __builtin_amdgcn_wmma_f32_16x16x32_bf16(
            false, af[mi], false, bf[ni], (short)0, acc[mi][ni], false, false);
  }

  // C/D layout: VGPR r -> M = r (+8 for lanes 16-31), N = lane&15.
  int rbase = (lane < 16) ? 0 : 8;
  #pragma unroll
  for (int mi = 0; mi < GEMM_MT; ++mi) {
    #pragma unroll
    for (int ni = 0; ni < GEMM_NT; ++ni) {
      int col = tnBase + ni * 16 + (lane & 15);
      #pragma unroll
      for (int r = 0; r < 8; ++r) {
        int m = tmBase + mi * 16 + rbase + r;
        float v = acc[mi][ni][r];
        if (addend != nullptr) v += addend[(size_t)m * ldc + col];
        C[(size_t)m * ldc + col] = v;
      }
    }
  }
}

// ---------------- conv (causal depthwise, time possibly reversed) + dt/a ----

__global__ __launch_bounds__(256) void conv_dt_kernel(
    const float* __restrict__ zx,     // [NTOK, D_IN_PAD], original time order
    const float* __restrict__ cw,     // [CONV_DIM, 4]
    const float* __restrict__ cb,     // [CONV_DIM]
    const float* __restrict__ dtbias, // [NHEADS]
    const float* __restrict__ Alog,   // [NHEADS]
    float* __restrict__ xbc,          // [NTOK, CONV_DIM]  (processed-domain order)
    float* __restrict__ dto,          // [NTOK, NHEADS]
    float* __restrict__ ao,           // [NTOK, NHEADS]
    int rev) {
  const int COLS = CONV_DIM + NHEADS;          // 1624
  int id = blockIdx.x * 256 + threadIdx.x;
  if (id >= B_SZ * SEQ * COLS) return;
  int b = id / (SEQ * COLS);
  int r = id % (SEQ * COLS);
  int t = r / COLS;
  int c = r % COLS;
  if (c < CONV_DIM) {
    float acc = cb[c];
    #pragma unroll
    for (int k = 0; k < D_CONV; ++k) {
      int s = t - (D_CONV - 1) + k;
      if (s >= 0) {
        int sm = rev ? (SEQ - 1 - s) : s;
        acc = fmaf(cw[c * D_CONV + k],
                   zx[(size_t)(b * SEQ + sm) * D_IN_PAD + D_INNER + c], acc);
      }
    }
    xbc[(size_t)(b * SEQ + t) * CONV_DIM + c] = acc * sigmoidf_(acc);
  } else {
    int h  = c - CONV_DIM;
    int tm = rev ? (SEQ - 1 - t) : t;
    float raw = zx[(size_t)(b * SEQ + tm) * D_IN_PAD + 2 * D_INNER + 2 * D_STATE + h]
                + dtbias[h];
    float dt = (raw > 20.f) ? raw : log1pf(expf(raw));   // softplus
    float a  = expf(-expf(Alog[h]) * dt);                // exp(dt * A), A=-exp(Alog)
    dto[(b * SEQ + t) * NHEADS + h] = dt;
    ao[(b * SEQ + t) * NHEADS + h] = a;
  }
}

// ---------------- selective scan: one wave per (batch, head) ----------------
// lane owns p = lane and p = lane+32; state h[p][n] in registers (2x32 floats).

__global__ __launch_bounds__(32) void scan_kernel(
    const float* __restrict__ xbc,   // [NTOK, CONV_DIM] processed-domain
    const float* __restrict__ dtb,   // [NTOK, NHEADS]
    const float* __restrict__ ab,    // [NTOK, NHEADS]
    const float* __restrict__ Dp,    // [NHEADS]
    float* __restrict__ y,           // [NTOK, D_INNER]   original time order
    int rev) {
  int b = blockIdx.x / NHEADS;
  int hh = blockIdx.x % NHEADS;
  int lane = threadIdx.x;
  float s0[D_STATE], s1[D_STATE];
  #pragma unroll
  for (int n = 0; n < D_STATE; ++n) { s0[n] = 0.f; s1[n] = 0.f; }
  float Dh = Dp[hh];

  for (int t = 0; t < SEQ; ++t) {
    int base = b * SEQ + t;
    float dt = dtb[base * NHEADS + hh];
    float a  = ab[base * NHEADS + hh];
    const float* row = xbc + (size_t)base * CONV_DIM;
    float x0 = row[hh * HEADDIM + lane];
    float x1 = row[hh * HEADDIM + 32 + lane];
    // B and C vectors (same addresses for all lanes -> broadcast)
    const float4* Bp = (const float4*)(row + D_INNER);
    const float4* Cp = (const float4*)(row + D_INNER + D_STATE);
    float Bl[D_STATE], Cl[D_STATE];
    #pragma unroll
    for (int q = 0; q < 8; ++q) {
      float4 bv = Bp[q], cv = Cp[q];
      Bl[4*q] = bv.x; Bl[4*q+1] = bv.y; Bl[4*q+2] = bv.z; Bl[4*q+3] = bv.w;
      Cl[4*q] = cv.x; Cl[4*q+1] = cv.y; Cl[4*q+2] = cv.z; Cl[4*q+3] = cv.w;
    }
    float dtx0 = dt * x0, dtx1 = dt * x1;
    float acc0 = Dh * x0, acc1 = Dh * x1;
    #pragma unroll
    for (int n = 0; n < D_STATE; ++n) {
      s0[n] = fmaf(a, s0[n], dtx0 * Bl[n]); acc0 = fmaf(s0[n], Cl[n], acc0);
      s1[n] = fmaf(a, s1[n], dtx1 * Bl[n]); acc1 = fmaf(s1[n], Cl[n], acc1);
    }
    int to = rev ? (SEQ - 1 - t) : t;
    float* yo = y + (size_t)(b * SEQ + to) * D_INNER + hh * HEADDIM;
    yo[lane] = acc0;
    yo[32 + lane] = acc1;
  }
}

// ---------------- gated RMSNorm: g = y*silu(z); out = rmsnorm(g)*nw (bf16) ---

__global__ __launch_bounds__(256) void gated_norm_kernel(
    const float* __restrict__ y,   // [NTOK, D_INNER]
    const float* __restrict__ zx,  // [NTOK, D_IN_PAD] (z = first D_INNER cols)
    const float* __restrict__ nw,  // [D_INNER]
    unsigned short* __restrict__ out) {
  __shared__ float g[D_INNER];
  __shared__ float red[32];
  int t = blockIdx.x;
  const float* yr = y + (size_t)t * D_INNER;
  const float* zr = zx + (size_t)t * D_IN_PAD;
  float ss = 0.f;
  for (int i = threadIdx.x; i < D_INNER; i += 256) {
    float z = zr[i];
    float gi = yr[i] * (z * sigmoidf_(z));
    g[i] = gi;
    ss += gi * gi;
  }
  ss = block_reduce_sum(ss, red);
  float r = rsqrtf(ss / (float)D_INNER + EPS);
  for (int i = threadIdx.x; i < D_INNER; i += 256)
    out[(size_t)t * D_INNER + i] = f2bf(g[i] * r * nw[i]);
}

// ---------------- concat(fwd, bwd) -> bf16 ----------------

__global__ __launch_bounds__(256) void concat_bf16_kernel(
    const float* __restrict__ f, const float* __restrict__ bw,
    unsigned short* __restrict__ out) {
  int id = blockIdx.x * 256 + threadIdx.x;
  if (id >= NTOK * 2 * D_MODEL) return;
  int t = id / (2 * D_MODEL), c = id % (2 * D_MODEL);
  float v = (c < D_MODEL) ? f[(size_t)t * D_MODEL + c]
                          : bw[(size_t)t * D_MODEL + (c - D_MODEL)];
  out[id] = f2bf(v);
}

// ---------------------------------------------------------------------------

extern "C" void kernel_launch(void* const* d_in, const int* in_sizes, int n_in,
                              void* d_out, int out_size, void* d_ws, size_t ws_size,
                              hipStream_t stream) {
  const float* x       = (const float*)d_in[0];
  const float* norm_w  = (const float*)d_in[1];
  const float* f_in_w  = (const float*)d_in[2];
  const float* f_cw    = (const float*)d_in[3];
  const float* f_cb    = (const float*)d_in[4];
  const float* f_dtb   = (const float*)d_in[5];
  const float* f_Alog  = (const float*)d_in[6];
  const float* f_D     = (const float*)d_in[7];
  const float* f_nw    = (const float*)d_in[8];
  const float* f_ow    = (const float*)d_in[9];
  const float* b_in_w  = (const float*)d_in[10];
  const float* b_cw    = (const float*)d_in[11];
  const float* b_cb    = (const float*)d_in[12];
  const float* b_dtb   = (const float*)d_in[13];
  const float* b_Alog  = (const float*)d_in[14];
  const float* b_D     = (const float*)d_in[15];
  const float* b_nw    = (const float*)d_in[16];
  const float* b_ow    = (const float*)d_in[17];
  const float* op_w    = (const float*)d_in[18];
  float* out           = (float*)d_out;

  // ---- workspace carve (256B aligned) ----
  size_t off = 0;
  auto alloc = [&](size_t bytes) -> void* {
    void* p = (char*)d_ws + off;
    off += (bytes + 255) & ~(size_t)255;
    return p;
  };
  unsigned short* hb   = (unsigned short*)alloc((size_t)NTOK * D_MODEL * 2);
  unsigned short* WiF  = (unsigned short*)alloc((size_t)D_IN_PAD * D_MODEL * 2);
  unsigned short* WiB  = (unsigned short*)alloc((size_t)D_IN_PAD * D_MODEL * 2);
  unsigned short* WoF  = (unsigned short*)alloc((size_t)D_MODEL * D_INNER * 2);
  unsigned short* WoB  = (unsigned short*)alloc((size_t)D_MODEL * D_INNER * 2);
  unsigned short* Wop  = (unsigned short*)alloc((size_t)D_MODEL * D_INNER * 2);
  float* zxF  = (float*)alloc((size_t)NTOK * D_IN_PAD * 4);
  float* zxB  = (float*)alloc((size_t)NTOK * D_IN_PAD * 4);
  float* xbcF = (float*)alloc((size_t)NTOK * CONV_DIM * 4);
  float* xbcB = (float*)alloc((size_t)NTOK * CONV_DIM * 4);
  float* dtF  = (float*)alloc((size_t)NTOK * NHEADS * 4);
  float* aF   = (float*)alloc((size_t)NTOK * NHEADS * 4);
  float* dtB  = (float*)alloc((size_t)NTOK * NHEADS * 4);
  float* aB   = (float*)alloc((size_t)NTOK * NHEADS * 4);
  float* yF   = (float*)alloc((size_t)NTOK * D_INNER * 4);
  float* yB   = (float*)alloc((size_t)NTOK * D_INNER * 4);
  unsigned short* gF = (unsigned short*)alloc((size_t)NTOK * D_INNER * 2);
  unsigned short* gB = (unsigned short*)alloc((size_t)NTOK * D_INNER * 2);
  float* foF  = (float*)alloc((size_t)NTOK * D_MODEL * 4);
  float* foB  = (float*)alloc((size_t)NTOK * D_MODEL * 4);
  unsigned short* cat = (unsigned short*)alloc((size_t)NTOK * 2 * D_MODEL * 2);
  (void)ws_size; (void)n_in; (void)in_sizes; (void)out_size;

  // ---- 1) input RMSNorm -> bf16 h ----
  rmsnorm_in_kernel<<<NTOK, 256, 0, stream>>>(x, norm_w, hb);

  // ---- 2) weight conversions ----
  {
    int n = D_IN_PAD * D_MODEL;
    cvt_bf16_pad_kernel<<<(n + 255) / 256, 256, 0, stream>>>(f_in_w, WiF, D_IN_PROJ, D_IN_PAD, D_MODEL);
    cvt_bf16_pad_kernel<<<(n + 255) / 256, 256, 0, stream>>>(b_in_w, WiB, D_IN_PROJ, D_IN_PAD, D_MODEL);
    int m = D_MODEL * D_INNER;
    cvt_bf16_kernel<<<(m + 255) / 256, 256, 0, stream>>>(f_ow, WoF, m);
    cvt_bf16_kernel<<<(m + 255) / 256, 256, 0, stream>>>(b_ow, WoB, m);
    cvt_bf16_kernel<<<(m + 255) / 256, 256, 0, stream>>>(op_w, Wop, m);
  }

  // ---- 3) in_proj GEMMs: [NTOK,768] x [3168,768]^T -> [NTOK,3168] ----
  {
    int blocks = (NTOK / 64) * (D_IN_PAD / 32);       // 16 * 99 = 1584 wave-blocks
    int grid = (blocks + 3) / 4;
    gemm_bf16_wmma_kernel<<<grid, 128, 0, stream>>>(
        (const bf16_t*)hb, D_MODEL, (const bf16_t*)WiF, D_MODEL,
        zxF, D_IN_PAD, nullptr, NTOK, D_IN_PAD, D_MODEL);
    gemm_bf16_wmma_kernel<<<grid, 128, 0, stream>>>(
        (const bf16_t*)hb, D_MODEL, (const bf16_t*)WiB, D_MODEL,
        zxB, D_IN_PAD, nullptr, NTOK, D_IN_PAD, D_MODEL);
  }

  // ---- 4) depthwise conv + SiLU + dt/a ----
  {
    int n = B_SZ * SEQ * (CONV_DIM + NHEADS);
    conv_dt_kernel<<<(n + 255) / 256, 256, 0, stream>>>(
        zxF, f_cw, f_cb, f_dtb, f_Alog, xbcF, dtF, aF, 0);
    conv_dt_kernel<<<(n + 255) / 256, 256, 0, stream>>>(
        zxB, b_cw, b_cb, b_dtb, b_Alog, xbcB, dtB, aB, 1);
  }

  // ---- 5) selective scan ----
  scan_kernel<<<B_SZ * NHEADS, 32, 0, stream>>>(xbcF, dtF, aF, f_D, yF, 0);
  scan_kernel<<<B_SZ * NHEADS, 32, 0, stream>>>(xbcB, dtB, aB, b_D, yB, 1);

  // ---- 6) gated RMSNorm -> bf16 ----
  gated_norm_kernel<<<NTOK, 256, 0, stream>>>(yF, zxF, f_nw, gF);
  gated_norm_kernel<<<NTOK, 256, 0, stream>>>(yB, zxB, b_nw, gB);

  // ---- 7) per-direction out_proj GEMMs: [NTOK,1536] x [768,1536]^T ----
  {
    int blocks = (NTOK / 64) * (D_MODEL / 32);        // 16 * 24 = 384
    int grid = (blocks + 3) / 4;
    gemm_bf16_wmma_kernel<<<grid, 128, 0, stream>>>(
        (const bf16_t*)gF, D_INNER, (const bf16_t*)WoF, D_INNER,
        foF, D_MODEL, nullptr, NTOK, D_MODEL, D_INNER);
    gemm_bf16_wmma_kernel<<<grid, 128, 0, stream>>>(
        (const bf16_t*)gB, D_INNER, (const bf16_t*)WoB, D_INNER,
        foB, D_MODEL, nullptr, NTOK, D_MODEL, D_INNER);
  }

  // ---- 8) concat + final projection with residual add ----
  {
    int n = NTOK * 2 * D_MODEL;
    concat_bf16_kernel<<<(n + 255) / 256, 256, 0, stream>>>(foF, foB, cat);
    int blocks = (NTOK / 64) * (D_MODEL / 32);
    int grid = (blocks + 3) / 4;
    gemm_bf16_wmma_kernel<<<grid, 128, 0, stream>>>(
        (const bf16_t*)cat, 2 * D_MODEL, (const bf16_t*)Wop, 2 * D_MODEL,
        out, D_MODEL, /*addend=*/x, NTOK, D_MODEL, 2 * D_MODEL);
  }
}
